// Self_Attn_GAN_16071767621668
// MI455X (gfx1250) — matrix operand order repaired and checked by reading, not compiled, not run on hardware
//
#include <hip/hip_runtime.h>
#include <hip/hip_bf16.h>

// Self-attention (GAN) fused kernels for gfx1250 (MI455X).
// B=4, C=256, H=W=64 -> N=4096, CQK=32.

#define B_    4
#define C_    256
#define N_    4096
#define CQK_  32
#define TN    128          // n-columns per projection workgroup
#define XT_LD 264          // padded bf16 row pitch for x^T tile in LDS (16B aligned)
#define STRIDE 4100        // padded f32 row pitch for energy strip in LDS

typedef __attribute__((ext_vector_type(16))) __bf16 v16bf;
typedef __attribute__((ext_vector_type(8)))  float  v8f;

union Frag { v16bf v; uint4 q[2]; };            // 32B WMMA A/B fragment
union H8   { uint4 q; __bf16 h[8]; };           // 16B packed bf16 octet

static __device__ inline unsigned pack2bf(float a, float b) {
    union { __bf16 h[2]; unsigned u; } r;
    r.h[0] = (__bf16)a; r.h[1] = (__bf16)b;
    return r.u;
}

static __device__ inline v8f wmma_bf16(v16bf a, v16bf b, v8f c) {
    return __builtin_amdgcn_wmma_f32_16x16x32_bf16(false, a, false, b,
                                                   (short)0, c, false, false);
}

// ---------------------------------------------------------------------------
// Kernel 1: fused Q/K/V 1x1-conv projections, f32 -> bf16 workspace.
//   Q  -> Qb [B][N][CQK]                (pair-packed b32 stores)
//   K  -> Ksw[B][N/16][lane32][16]      (fragment-native, 1x b128 store/tile)
//   V  -> Vsw[B][N/32][C/16][lane32][16](fragment-native, 1x b128 store/tile,
//                                        computed transposed: A=x^T, B=wv^T)
// grid = (N/TN, B), block = 256 (8 wave32)
// ---------------------------------------------------------------------------
extern "C" __global__ __launch_bounds__(256)
void qkv_proj_kernel(const float* __restrict__ x,
                     const float* __restrict__ wq, const float* __restrict__ bq,
                     const float* __restrict__ wk, const float* __restrict__ bk,
                     const float* __restrict__ wv, const float* __restrict__ bv,
                     __bf16* __restrict__ Qb,
                     __bf16* __restrict__ Ksw,
                     __bf16* __restrict__ Vsw)
{
    extern __shared__ __bf16 xt[];              // x^T tile: [TN][XT_LD] (~66 KB)
    const int n0  = blockIdx.x * TN;
    const int b   = blockIdx.y;
    const int tid = threadIdx.x;

    // Stage x^T into LDS: float4 global reads (coalesced), bf16 scatter stores.
    for (int idx = tid; idx < C_ * TN / 4; idx += 256) {
        const int c  = idx >> 5;                // 0..255
        const int n4 = (idx & 31) * 4;          // 0..124
        const float4 v = *(const float4*)(x + ((size_t)b * C_ + c) * N_ + n0 + n4);
        xt[(n4 + 0) * XT_LD + c] = (__bf16)v.x;
        xt[(n4 + 1) * XT_LD + c] = (__bf16)v.y;
        xt[(n4 + 2) * XT_LD + c] = (__bf16)v.z;
        xt[(n4 + 3) * XT_LD + c] = (__bf16)v.w;
    }
    __syncthreads();

    const int wave = tid >> 5;
    const int lane = tid & 31;
    const int lhi  = lane >> 4;
    const int l16  = lane & 15;

    // ---- Q and K tiles: A = weights (M=oc), B = x^T columns from LDS.
    for (int p = 0; p < 2; ++p) {
        const float* w   = p ? wk : wq;
        const float* bia = p ? bk : bq;
        #pragma unroll
        for (int i = 0; i < 2; ++i) {
            const int tt  = wave + 8 * i;       // 0..15
            const int ct  = tt & 7;
            const int oc0 = (tt >> 3) * 16;
            const int cn0 = ct * 16;

            v8f acc;
            #pragma unroll
            for (int r = 0; r < 8; ++r) acc[r] = bia[oc0 + r + lhi * 8];

            #pragma unroll
            for (int kk = 0; kk < C_; kk += 32) {
                const float* wrow = w + (size_t)(oc0 + l16) * C_ + kk + lhi * 16;
                v16bf a;
                #pragma unroll
                for (int j = 0; j < 8; ++j) {
                    a[j]     = (__bf16)wrow[j];
                    a[8 + j] = (__bf16)wrow[8 + j];
                }
                Frag fb;                         // 32B contiguous -> 2x ds_load_b128
                const uint4* bp = (const uint4*)(xt + (cn0 + l16) * XT_LD + kk + lhi * 16);
                fb.q[0] = bp[0]; fb.q[1] = bp[1];
                acc = wmma_bf16(a, fb.v, acc);
            }

            const int n = n0 + cn0 + l16;
            if (p == 0) {                        // Q: 4x packed b32 stores
                unsigned* qp = (unsigned*)(Qb + ((size_t)b * N_ + n) * CQK_ + lhi * 8);
                #pragma unroll
                for (int r = 0; r < 4; ++r) qp[r] = pack2bf(acc[2 * r], acc[2 * r + 1]);
            } else {                             // K: one b128 store into Ksw
                H8 st;
                #pragma unroll
                for (int r = 0; r < 8; ++r) st.h[r] = (__bf16)acc[r];
                const int ctg = (n0 >> 4) + ct;
                const int lp  = l16 + ((oc0 >> 4) << 4);
                ((uint4*)Ksw)[(((size_t)b * 256 + ctg) * 32 + lp) * 2 + lhi] = st.q;
            }
        }
    }

    // ---- V tiles, transposed: A = x^T rows (M=n), B = wv^T cols (N=cout).
    for (int i = 0; i < 16; ++i) {
        const int tt  = wave + 8 * i;            // 0..127
        const int ct  = tt & 7;                  // n tile
        const int oc0 = (tt >> 3) * 16;          // channel tile
        const int cn0 = ct * 16;

        const float bv0 = bv[oc0 + l16];         // D col = oc0+l16, same for all r
        v8f acc;
        #pragma unroll
        for (int r = 0; r < 8; ++r) acc[r] = bv0;

        #pragma unroll
        for (int kk = 0; kk < C_; kk += 32) {
            Frag fa;                             // A: two 16B ds loads
            fa.q[0] = *(const uint4*)(xt + (cn0 + l16) * XT_LD + kk + lhi * 16);
            fa.q[1] = *(const uint4*)(xt + (cn0 + l16) * XT_LD + kk + lhi * 16 + 8);
            const float* wrow = wv + (size_t)(oc0 + l16) * C_ + kk + lhi * 16;
            v16bf bw;                            // B: 16 consecutive f32 -> cvt
            #pragma unroll
            for (int j = 0; j < 16; ++j) bw[j] = (__bf16)wrow[j];
            acc = wmma_bf16(fa.v, bw, acc);
        }

        H8 st;
        #pragma unroll
        for (int r = 0; r < 8; ++r) st.h[r] = (__bf16)acc[r];
        const int nblk = (n0 + cn0) >> 5;
        const int cblk = oc0 >> 4;
        const int lp   = l16 + ((ct & 1) << 4);
        ((uint4*)Vsw)[((((size_t)b * 128 + nblk) * 16 + cblk) * 32 + lp) * 2 + lhi] = st.q;
    }
}

// ---------------------------------------------------------------------------
// Kernel 2: fused attention for one 16-query-row strip (262 KB LDS strip).
// grid = (N/16, B), block = 256 (8 wave32)
// ---------------------------------------------------------------------------
extern "C" __global__ __launch_bounds__(256)
void attn_fused_kernel(const float* __restrict__ x,
                       const __bf16* __restrict__ Qb,
                       const __bf16* __restrict__ Ksw,
                       const __bf16* __restrict__ Vsw,
                       const float* __restrict__ gamma,
                       float* __restrict__ out,     // [B][C][N]
                       float* __restrict__ attn)    // [B][N][N]
{
    extern __shared__ float strip[];           // [16][STRIDE] + red[256] + red2[16]
    float* red  = strip + 16 * STRIDE;
    float* red2 = red + 256;

    const int n0   = blockIdx.x * 16;
    const int b    = blockIdx.y;
    const int tid  = threadIdx.x;
    const int wave = tid >> 5;
    const int lane = tid & 31;
    const int lhi  = lane >> 4;
    const int l16  = lane & 15;

    // Q fragment for this strip (two 16B loads, reused by 32 energy tiles).
    Frag aq;
    {
        const __bf16* qrow = Qb + ((size_t)b * N_ + n0 + l16) * CQK_;
        aq.q[0] = *(const uint4*)(qrow + lhi * 16);
        aq.q[1] = *(const uint4*)(qrow + lhi * 16 + 8);
    }

    // ---- Phase A: energy strip. Wave w owns key columns [w*512, w*512+512).
    for (int ct = 0; ct < 32; ++ct) {
        const int c0 = wave * 512 + ct * 16;
        const int cg = c0 >> 4;
        const uint4* kp = (const uint4*)Ksw + (((size_t)b * 256 + cg) * 32 + lane) * 2;
        Frag fb; fb.q[0] = kp[0]; fb.q[1] = kp[1];
        v8f e = {};
        e = wmma_bf16(aq.v, fb.v, e);
        #pragma unroll
        for (int r = 0; r < 8; ++r)
            strip[(r + lhi * 8) * STRIDE + c0 + l16] = e[r];
    }
    __syncthreads();

    // ---- Phase B: softmax statistics (16 threads per row).
    const int rr = tid >> 4;
    const int cc = tid & 15;
    float m = -3.4e38f;
    for (int c = cc; c < N_; c += 16) m = fmaxf(m, strip[rr * STRIDE + c]);
    red[rr * 16 + cc] = m;
    __syncthreads();
    if (cc == 0) {
        float mm = red[rr * 16];
        #pragma unroll
        for (int j = 1; j < 16; ++j) mm = fmaxf(mm, red[rr * 16 + j]);
        red2[rr] = mm;
    }
    __syncthreads();
    const float rowm = red2[rr];
    float s = 0.f;
    for (int c = cc; c < N_; c += 16) {
        const float v = __expf(strip[rr * STRIDE + c] - rowm);
        strip[rr * STRIDE + c] = v;
        s += v;
    }
    red[rr * 16 + cc] = s;
    __syncthreads();
    if (cc == 0) {
        float ss = 0.f;
        #pragma unroll
        for (int j = 0; j < 16; ++j) ss += red[rr * 16 + j];
        red2[rr] = 1.0f / ss;
    }
    __syncthreads();

    // ---- Phase C: normalize in place + single coalesced attn write to HBM.
    for (int idx = tid; idx < 16 * N_; idx += 256) {
        const int r = idx >> 12;
        const int c = idx & (N_ - 1);
        const float v = strip[r * STRIDE + c] * red2[r];
        strip[r * STRIDE + c] = v;
        attn[((size_t)b * N_ + n0 + r) * N_ + c] = v;
    }
    __syncthreads();

    // ---- Phase D: O = attn @ V. Wave w owns channels [w*32, w*32+32).
    const int c0 = wave * 32;
    const int cb = c0 >> 4;
    v8f o0 = {}, o1 = {};
    for (int kk = 0; kk < N_; kk += 32) {
        // A fragment: 16 aligned f32 from the strip -> bf16 (v_cvt_pk_bf16_f32).
        const float4* ap  = (const float4*)(strip + l16 * STRIDE + kk + lhi * 16);
        const float4* ap2 = (const float4*)(strip + l16 * STRIDE + kk + lhi * 16 + 8);
        const float4 f0 = ap[0],  f1 = ap[1];
        const float4 f2 = ap2[0], f3 = ap2[1];
        v16bf aa;
        aa[0]  = (__bf16)f0.x; aa[1]  = (__bf16)f0.y; aa[2]  = (__bf16)f0.z; aa[3]  = (__bf16)f0.w;
        aa[4]  = (__bf16)f1.x; aa[5]  = (__bf16)f1.y; aa[6]  = (__bf16)f1.z; aa[7]  = (__bf16)f1.w;
        aa[8]  = (__bf16)f2.x; aa[9]  = (__bf16)f2.y; aa[10] = (__bf16)f2.z; aa[11] = (__bf16)f2.w;
        aa[12] = (__bf16)f3.x; aa[13] = (__bf16)f3.y; aa[14] = (__bf16)f3.z; aa[15] = (__bf16)f3.w;

        // B fragments: fragment-native Vsw, 2x b128 each, coalesced.
        const uint4* vp = (const uint4*)Vsw +
            ((((size_t)b * 128 + (kk >> 5)) * 16 + cb) * 32 + lane) * 2;
        Frag b0, b1;
        b0.q[0] = vp[0];      b0.q[1] = vp[1];
        b1.q[0] = vp[64];     b1.q[1] = vp[65];      // next channel tile (+32 lanes)

        o0 = wmma_bf16(aa, b0.v, o0);
        o1 = wmma_bf16(aa, b1.v, o1);
    }

    // Epilogue: out[b,c,n] = x[b,c,n] + gamma * o[b,n,c] (fused transpose).
    const float g = gamma[0];
    #pragma unroll
    for (int r = 0; r < 8; ++r) {
        const int nq = n0 + r + lhi * 8;
        const int ch = c0 + l16;
        const size_t i0 = ((size_t)b * C_ + ch) * N_ + nq;
        out[i0] = x[i0] + g * o0[r];
        const size_t i1 = ((size_t)b * C_ + ch + 16) * N_ + nq;
        out[i1] = x[i1] + g * o1[r];
    }
}

// ---------------------------------------------------------------------------
extern "C" void kernel_launch(void* const* d_in, const int* in_sizes, int n_in,
                              void* d_out, int out_size, void* d_ws, size_t ws_size,
                              hipStream_t stream) {
    (void)in_sizes; (void)n_in; (void)out_size; (void)ws_size;
    const float* x     = (const float*)d_in[0];
    const float* wq    = (const float*)d_in[1];
    const float* bq    = (const float*)d_in[2];
    const float* wk    = (const float*)d_in[3];
    const float* bk    = (const float*)d_in[4];
    const float* wv    = (const float*)d_in[5];
    const float* bv    = (const float*)d_in[6];
    const float* gamma = (const float*)d_in[7];

    float* out  = (float*)d_out;
    float* attn = out + (size_t)B_ * C_ * N_;   // tuple: (out, attn) flat

    // Workspace: bf16 Q (1 MB) | bf16 K swizzled (1 MB) | bf16 V swizzled (8 MB)
    __bf16* Qb  = (__bf16*)d_ws;
    __bf16* Ksw = Qb + (size_t)B_ * N_ * CQK_;
    __bf16* Vsw = Ksw + (size_t)B_ * N_ * CQK_;

    const size_t lds1 = (size_t)TN * XT_LD * sizeof(__bf16);              // ~66 KB
    const size_t lds2 = (size_t)(16 * STRIDE + 256 + 16) * sizeof(float); // ~263 KB
    (void)hipFuncSetAttribute((const void*)qkv_proj_kernel,
                              hipFuncAttributeMaxDynamicSharedMemorySize, (int)lds1);
    (void)hipFuncSetAttribute((const void*)attn_fused_kernel,
                              hipFuncAttributeMaxDynamicSharedMemorySize, (int)lds2);

    dim3 blk(256);
    dim3 g1(N_ / TN, B_);
    qkv_proj_kernel<<<g1, blk, lds1, stream>>>(x, wq, bq, wk, bk, wv, bv, Qb, Ksw, Vsw);

    dim3 g2(N_ / 16, B_);
    attn_fused_kernel<<<g2, blk, lds2, stream>>>(x, Qb, Ksw, Vsw, gamma, out, attn);
}
